// LigandEncoder_4097398800930
// MI455X (gfx1250) — compile-verified, matrix operand order checked by reading, not run on hardware
//
#include <hip/hip_runtime.h>
#include <hip/hip_bf16.h>

// ---------------------------------------------------------------------------
// GIN (3 layers, D=H=128) + global_add_pool for MI455X (gfx1250, wave32).
//  - scatter/gather phase: L2-resident, native f32 atomics (no CAS),
//    global_prefetch to hide gather latency
//  - MLP phase: bf16 v_wmma_f32_16x16x32_bf16, f32 accumulate, LDS-tiled,
//    weights staged with GLOBAL_LOAD_ASYNC_TO_LDS_B128 (ASYNCcnt path)
// ---------------------------------------------------------------------------

typedef __attribute__((ext_vector_type(16))) __bf16 v16bf;
typedef __attribute__((ext_vector_type(8)))  __bf16 v8bf;
typedef __attribute__((ext_vector_type(8)))  float  v8f;
typedef int gv4i __attribute__((vector_size(16)));   // int4 for async-LDS builtin

#define HD 128   // feature dim (D == H == 128)

#define AS1 __attribute__((address_space(1)))
#define AS3 __attribute__((address_space(3)))

#if __has_builtin(__builtin_amdgcn_global_load_async_to_lds_b128) && \
    __has_builtin(__builtin_amdgcn_s_wait_asynccnt)
#define USE_ASYNC_LDS 1
#endif

__device__ __forceinline__ void atomAddF32(float* p, float v) {
#if __has_builtin(__builtin_amdgcn_global_atomic_fadd_f32)
    // fire-and-forget hardware atomic: global_atomic_add_f32 (STOREcnt path)
    __builtin_amdgcn_global_atomic_fadd_f32((AS1 float*)(uintptr_t)p, v);
#else
    __hip_atomic_fetch_add(p, v, __ATOMIC_RELAXED, __HIP_MEMORY_SCOPE_AGENT);
#endif
}

// ---- weight convert: W[k][n] f32 -> Wt[n][k] bf16 (transposed for B-frags) --
__global__ void convert_w(const float* __restrict__ W, __bf16* __restrict__ Wt) {
    int i = blockIdx.x * blockDim.x + threadIdx.x;
    if (i < HD * HD) {
        int k = i >> 7, n = i & 127;
        Wt[n * HD + k] = (__bf16)W[i];
    }
}

// ---- edge scatter-add: agg[dst] += h[src], one wave per edge ---------------
__global__ void gin_scatter(const float* __restrict__ h,
                            const int* __restrict__ src,
                            const int* __restrict__ dst,
                            float* __restrict__ agg, int nEdges) {
    const int lane   = threadIdx.x & 31;
    const int wave   = (blockIdx.x * blockDim.x + threadIdx.x) >> 5;
    const int nWaves = (gridDim.x * blockDim.x) >> 5;
    for (int e = wave; e < nEdges; e += nWaves) {
        const int s = src[e];
        const int d = dst[e];
        // prefetch the next edge's source row while this one drains atomics
        const int e2 = e + nWaves;
        if (e2 < nEdges)
            __builtin_prefetch(h + (size_t)src[e2] * HD + lane * 4, 0, 1);
        const float4 v = *reinterpret_cast<const float4*>(h + (size_t)s * HD + lane * 4);
        float* p = agg + (size_t)d * HD + lane * 4;
        atomAddF32(p + 0, v.x);
        atomAddF32(p + 1, v.y);
        atomAddF32(p + 2, v.z);
        atomAddF32(p + 3, v.w);
    }
}

// ---- fused GIN MLP: hout = relu(relu((agg+hin)W1+b1)W2+b2) -----------------
// 128-row tile per block, 8 waves x 16 rows, bf16 WMMA, f32 accum.
__global__ __launch_bounds__(256) void gin_mlp(
        const float*  __restrict__ agg,  const float* __restrict__ hin,
        const __bf16* __restrict__ w1t,  const float* __restrict__ b1,
        const __bf16* __restrict__ w2t,  const float* __restrict__ b2,
        float* __restrict__ hout, int nNodes) {
    __shared__ __align__(16) __bf16 sW1[HD * HD];   // W1^T  [n][k]
    __shared__ __align__(16) __bf16 sW2[HD * HD];   // W2^T  [n][k]
    __shared__ __align__(16) __bf16 sA [HD * HD];   // A tile / intermediate t

    const int tid     = threadIdx.x;
    const int rowBase = blockIdx.x * HD;

    // stage transposed bf16 weights (32 KB each) -> LDS
#ifdef USE_ASYNC_LDS
    // async DMA path: per-lane 16B global->LDS, bypasses VGPRs (ASYNCcnt)
    for (int i = tid; i < (HD * HD) / 8; i += 256) {
        __builtin_amdgcn_global_load_async_to_lds_b128(
            (AS1 gv4i*)(uintptr_t)(w1t + (size_t)i * 8),
            (AS3 gv4i*)(AS3 void*)(sW1 + (size_t)i * 8), 0, 0);
        __builtin_amdgcn_global_load_async_to_lds_b128(
            (AS1 gv4i*)(uintptr_t)(w2t + (size_t)i * 8),
            (AS3 gv4i*)(AS3 void*)(sW2 + (size_t)i * 8), 0, 0);
    }
#else
    for (int i = tid; i < (HD * HD) / 8; i += 256) {
        reinterpret_cast<ulonglong2*>(sW1)[i] = reinterpret_cast<const ulonglong2*>(w1t)[i];
        reinterpret_cast<ulonglong2*>(sW2)[i] = reinterpret_cast<const ulonglong2*>(w2t)[i];
    }
#endif
    // stage A = agg + hin (f32 -> bf16)
    for (int i = tid; i < HD * (HD / 4); i += 256) {       // 4096 float4 slots
        const int r  = i >> 5;                              // row in tile
        const int c4 = i & 31;                              // float4 column
        const long gr = (long)rowBase + r;
        float4 a = make_float4(0.f, 0.f, 0.f, 0.f);
        if (gr < nNodes) {
            const float4 va = *reinterpret_cast<const float4*>(agg + gr * HD + c4 * 4);
            const float4 vh = *reinterpret_cast<const float4*>(hin + gr * HD + c4 * 4);
            a.x = va.x + vh.x; a.y = va.y + vh.y; a.z = va.z + vh.z; a.w = va.w + vh.w;
        }
        __bf16* p = sA + r * HD + c4 * 4;
        p[0] = (__bf16)a.x; p[1] = (__bf16)a.y; p[2] = (__bf16)a.z; p[3] = (__bf16)a.w;
    }
#ifdef USE_ASYNC_LDS
    __builtin_amdgcn_s_wait_asynccnt(0);   // drain async weight DMA
#endif
    __syncthreads();

    const int wave = tid >> 5;
    const int lane = tid & 31;
    const int half = lane >> 4;       // lane group: K 0-7/16-23 vs 8-15/24-31
    const int ln   = lane & 15;       // M (A-frag) / N (B-frag) within tile
    const int kb   = half * 8;
    const int mrow = wave * 16;       // wave's 16-row strip inside the tile

    // 16-bit A/B fragment layout: two contiguous 8-element (16 B) runs per lane
    auto loadFragA = [&](const __bf16* base, int k0) -> v16bf {
        const v8bf lo = *reinterpret_cast<const v8bf*>(base + (mrow + ln) * HD + k0 + kb);
        const v8bf hi = *reinterpret_cast<const v8bf*>(base + (mrow + ln) * HD + k0 + kb + 16);
        return __builtin_shufflevector(lo, hi, 0,1,2,3,4,5,6,7,8,9,10,11,12,13,14,15);
    };
    auto loadFragB = [&](const __bf16* wt, int ntile, int k0) -> v16bf {
        const v8bf lo = *reinterpret_cast<const v8bf*>(wt + (ntile * 16 + ln) * HD + k0 + kb);
        const v8bf hi = *reinterpret_cast<const v8bf*>(wt + (ntile * 16 + ln) * HD + k0 + kb + 16);
        return __builtin_shufflevector(lo, hi, 0,1,2,3,4,5,6,7,8,9,10,11,12,13,14,15);
    };

    v8f acc[8];
    #pragma unroll
    for (int t = 0; t < 8; ++t) acc[t] = v8f{};

    // ---- GEMM 1: t = (agg+h) @ W1 ----
    #pragma unroll
    for (int k0 = 0; k0 < HD; k0 += 32) {
        const v16bf a = loadFragA(sA, k0);
        #pragma unroll
        for (int t = 0; t < 8; ++t) {
            const v16bf b = loadFragB(sW1, t, k0);
            acc[t] = __builtin_amdgcn_wmma_f32_16x16x32_bf16(
                         false, a, false, b, (short)0, acc[t], false, false);
        }
    }
    // bias + ReLU, write intermediate back into this wave's own LDS rows
    #pragma unroll
    for (int t = 0; t < 8; ++t) {
        const int   n  = t * 16 + ln;
        const float bb = b1[n];
        #pragma unroll
        for (int e = 0; e < 8; ++e) {
            const float v = fmaxf(acc[t][e] + bb, 0.f);
            sA[(mrow + half * 8 + e) * HD + n] = (__bf16)v;
        }
        acc[t] = v8f{};
    }
    __syncthreads();   // cheap safety for the LDS write->read turnaround

    // ---- GEMM 2: z = t @ W2 ----
    #pragma unroll
    for (int k0 = 0; k0 < HD; k0 += 32) {
        const v16bf a = loadFragA(sA, k0);
        #pragma unroll
        for (int t = 0; t < 8; ++t) {
            const v16bf b = loadFragB(sW2, t, k0);
            acc[t] = __builtin_amdgcn_wmma_f32_16x16x32_bf16(
                         false, a, false, b, (short)0, acc[t], false, false);
        }
    }
    // bias + outer ReLU -> f32 h_out
    #pragma unroll
    for (int t = 0; t < 8; ++t) {
        const int   n  = t * 16 + ln;
        const float bb = b2[n];
        #pragma unroll
        for (int e = 0; e < 8; ++e) {
            const long row = (long)rowBase + mrow + half * 8 + e;
            if (row < nNodes)
                hout[row * HD + n] = fmaxf(acc[t][e] + bb, 0.f);
        }
    }
}

// ---- global_add_pool: out[batch[v]] += h[v], one wave per node -------------
__global__ void gin_pool(const float* __restrict__ h, const int* __restrict__ batch,
                         float* __restrict__ out, int nNodes) {
    const int lane   = threadIdx.x & 31;
    const int wave   = (blockIdx.x * blockDim.x + threadIdx.x) >> 5;
    const int nWaves = (gridDim.x * blockDim.x) >> 5;
    for (int v = wave; v < nNodes; v += nWaves) {
        const int g = batch[v];
        const float4 val = *reinterpret_cast<const float4*>(h + (size_t)v * HD + lane * 4);
        float* p = out + (size_t)g * HD + lane * 4;
        atomAddF32(p + 0, val.x);
        atomAddF32(p + 1, val.y);
        atomAddF32(p + 2, val.z);
        atomAddF32(p + 3, val.w);
    }
}

extern "C" void kernel_launch(void* const* d_in, const int* in_sizes, int n_in,
                              void* d_out, int out_size, void* d_ws, size_t ws_size,
                              hipStream_t stream) {
    const float* x     = (const float*)d_in[0];
    const int*   ei    = (const int*)d_in[1];
    const int*   batch = (const int*)d_in[2];

    const int N = in_sizes[0] / HD;     // nodes
    const int E = in_sizes[1] / 2;      // edges
    const int* src = ei;
    const int* dst = ei + E;

    // workspace layout: agg | hA | hB | 6x bf16 transposed weights
    char*  ws = (char*)d_ws;
    const size_t fN = (size_t)N * HD;
    float*  agg = (float*)(ws);
    float*  hA  = (float*)(ws + fN * 4);
    float*  hB  = (float*)(ws + fN * 8);
    __bf16* wt  = (__bf16*)(ws + fN * 12);          // 6 * 128*128 bf16

    // one-time (per launch) weight convert+transpose to bf16
    for (int l = 0; l < 3; ++l) {
        const float* W1 = (const float*)d_in[3 + 4 * l + 0];
        const float* W2 = (const float*)d_in[3 + 4 * l + 2];
        convert_w<<<(HD * HD + 255) / 256, 256, 0, stream>>>(W1, wt + (2 * l + 0) * HD * HD);
        convert_w<<<(HD * HD + 255) / 256, 256, 0, stream>>>(W2, wt + (2 * l + 1) * HD * HD);
    }

    const float* hin = x;
    float* bufs[3] = { hA, hB, hA };
    for (int l = 0; l < 3; ++l) {
        const float* b1 = (const float*)d_in[3 + 4 * l + 1];
        const float* b2 = (const float*)d_in[3 + 4 * l + 3];
        (void)hipMemsetAsync(agg, 0, fN * sizeof(float), stream);
        gin_scatter<<<8192, 256, 0, stream>>>(hin, src, dst, agg, E);
        gin_mlp<<<(N + HD - 1) / HD, 256, 0, stream>>>(
            agg, hin,
            wt + (2 * l + 0) * HD * HD, b1,
            wt + (2 * l + 1) * HD * HD, b2,
            bufs[l], N);
        hin = bufs[l];
    }

    (void)hipMemsetAsync(d_out, 0, (size_t)out_size * sizeof(float), stream);
    gin_pool<<<2048, 256, 0, stream>>>(hin, batch, (float*)d_out, N);
}
